// unit_gcn_36661840838902
// MI455X (gfx1250) — compile-verified, matrix operand order checked by reading, not run on hardware
//
#include <hip/hip_runtime.h>
#include <stdint.h>

#define N_ 32
#define C_ 64
#define T_ 256
#define V_ 25
#define O_ 192
#define S_ 3
#define R_ 8
#define TB 16
#define NT ((TB * V_) / 16) /* 25 column tiles of 16 per t-block */

typedef __attribute__((ext_vector_type(16))) __bf16 v16bf;
typedef __attribute__((ext_vector_type(8))) float v8f;

union Frag16 { v16bf v; uint32_t u[8]; };
union Acc8   { v8f v;   float f[8];   };

// native f32<->bf16 conversion (lowers to v_cvt_* on gfx1250, RNE)
__device__ __forceinline__ uint16_t f2bf(float x) {
  __bf16 b = (__bf16)x;
  return __builtin_bit_cast(uint16_t, b);
}
__device__ __forceinline__ float bf2f(uint16_t h) {
  return (float)__builtin_bit_cast(__bf16, h);
}
// CDNA5 16-bit WMMA operand layout: VGPR j of lane-half h holds K pair starting at:
__device__ __forceinline__ int kbase(int j, int h) {
  return (j < 4 ? 2 * j : 16 + 2 * (j - 4)) + 8 * h;
}

// ---------------- prep kernels ----------------

// x (N,C,T,V) f32 -> xT (N,T,V,C) bf16  (c contiguous so WMMA k-pairs are one b32 load)
__global__ void k_cvt_xT(const float* __restrict__ x, uint16_t* __restrict__ xT) {
  int i = blockIdx.x * 256 + threadIdx.x;
  if (i >= N_ * C_ * T_ * V_) return;
  int v = i % V_, t = (i / V_) % T_, c = (i / (V_ * T_)) % C_, n = i / (V_ * T_ * C_);
  xT[(((size_t)n * T_ + t) * V_ + v) * C_ + c] = f2bf(x[i]);
}

// xm (N,C,V) = mean over T
__global__ void k_mean_t(const float* __restrict__ x, float* __restrict__ xm) {
  int i = blockIdx.x * 256 + threadIdx.x;
  if (i >= N_ * C_ * V_) return;
  int v = i % V_, nc = i / V_;
  const float* p = x + (size_t)nc * T_ * V_ + v;
  float s = 0.f;
  for (int t = 0; t < T_; ++t) s += p[(size_t)t * V_];
  xm[i] = s * (1.0f / T_);
}

// x1/x2 (N,S,R,V) tiny projections of xm
__global__ void k_x12(const float* __restrict__ xm, const float* __restrict__ w1,
                      const float* __restrict__ b1, const float* __restrict__ w2,
                      const float* __restrict__ b2, float* __restrict__ x1,
                      float* __restrict__ x2) {
  int i = blockIdx.x * 256 + threadIdx.x;
  const int total = N_ * S_ * R_ * V_;
  if (i >= 2 * total) return;
  int which = i / total, j = i % total;
  int v = j % V_, r = (j / V_) % R_, s = (j / (V_ * R_)) % S_, n = j / (V_ * R_ * S_);
  const float* w = (which ? w2 : w1) + ((size_t)s * R_ + r) * C_;
  const float* xp = xm + (size_t)n * C_ * V_ + v;
  float acc = (which ? b2 : b1)[s * R_ + r];
  for (int c = 0; c < C_; ++c) acc += w[c] * xp[(size_t)c * V_];
  (which ? x2 : x1)[j] = acc;
}

// d (N,S,R,V,V) = tanh(x1[u] - x2[v])  -- computed once, reused by all 192 o's
__global__ void k_dtanh(const float* __restrict__ x1, const float* __restrict__ x2,
                        float* __restrict__ dt) {
  int i = blockIdx.x * 256 + threadIdx.x;
  if (i >= N_ * S_ * R_ * V_ * V_) return;
  int v = i % V_, u = (i / V_) % V_, nsr = i / (V_ * V_);
  dt[i] = tanhf(x1[nsr * V_ + u] - x2[nsr * V_ + v]);
}

// pack conv3_w (s=0..2) and down_w (as s=3) -> bf16 [4][O][C]
__global__ void k_wpack(const float* __restrict__ c3w, const float* __restrict__ dw,
                        uint16_t* __restrict__ wbf) {
  int i = blockIdx.x * 256 + threadIdx.x;
  if (i >= 4 * O_ * C_) return;
  int c = i % C_, o = (i / C_) % O_, s = i / (C_ * O_);
  float w = (s < S_) ? c3w[((size_t)s * O_ + o) * C_ + c] : dw[(size_t)o * C_ + c];
  wbf[i] = f2bf(w);
}

// A_at -> bf16 [n][s][o][u:32][v:32] (zero padded) + ybias[n][s][o][u] = b3 * rowsum
__global__ void k_abf(const float* __restrict__ dt, const float* __restrict__ w4,
                      const float* __restrict__ b4, const float* __restrict__ A3,
                      const float* __restrict__ A6, const float* __restrict__ spd,
                      const float* __restrict__ alpha, const float* __restrict__ beta,
                      const float* __restrict__ gamma, const float* __restrict__ c3b,
                      uint16_t* __restrict__ abf, float* __restrict__ ybias) {
  __shared__ float sh_a[32 * 32];
  int blk = blockIdx.x;                         // (n*S+s)*O + o
  int o = blk % O_, s = (blk / O_) % S_, n = blk / (O_ * S_);
  float al = alpha[0], be = beta[0], ga = gamma[0];
  float w4r[R_];
#pragma unroll
  for (int r = 0; r < R_; ++r) w4r[r] = w4[((size_t)s * O_ + o) * R_ + r];
  float bb = b4[s * O_ + o];
  const float* dts = dt + (size_t)(n * S_ + s) * (R_ * V_ * V_);
  uint16_t* out = abf + (size_t)blk * 32 * 32;
  for (int idx = threadIdx.x; idx < 32 * 32; idx += blockDim.x) {
    int u = idx >> 5, v = idx & 31;
    float a = 0.f;
    if (u < V_ && v < V_) {
      float adyn = bb;
#pragma unroll
      for (int r = 0; r < R_; ++r) adyn += w4r[r] * dts[(r * V_ + u) * V_ + v];
      a = al * adyn + A3[(s * V_ + u) * V_ + v] + be * A6[((o % 6) * V_ + u) * V_ + v]
          + ga * spd[u * V_ + v];
    }
    sh_a[idx] = a;
    out[idx] = f2bf(a);
  }
  __syncthreads();
  float b3 = c3b[s * O_ + o];
  if (threadIdx.x < V_) {
    int u = threadIdx.x;
    float rs = 0.f;
    for (int v = 0; v < V_; ++v) rs += sh_a[u * 32 + v];
    ybias[(size_t)blk * 32 + u] = b3 * rs;
  } else if (threadIdx.x < 32) {
    ybias[(size_t)blk * 32 + threadIdx.x] = 0.f;
  }
}

// ---------------- main fused WMMA kernel ----------------
// grid (O/16, N); block 256 = 8 waves.
// Stage A: D[o16 x tv16] = W(M=o,K=c64) x X(K=c,N=tv)   (s=0..2 -> x3 LDS, s=3 -> xd)
// Stage B: D[t16 x u16]  = X3(M=t,K=v32) x A^T(K=v,N=u), accumulated over s.
__global__ void __launch_bounds__(256) k_main(
    const uint16_t* __restrict__ xT,     // [N][T][V][C] bf16 (L2-resident, 26 MB)
    const uint16_t* __restrict__ wbf,    // [4][O][C]    bf16
    const uint16_t* __restrict__ abf,    // [N][S][O][32][32] bf16 (u,v), padded
    const float* __restrict__ ybias,     // [N][S][O][32]
    const float* __restrict__ down_b,    // [O]
    uint16_t* __restrict__ xdbf,         // [N][O][T][V] bf16
    float* __restrict__ yraw)            // [N][O][T][V] f32 (= d_out, raw pre-BN)
{
  __shared__ uint16_t x3[S_][16][TB][32];        // 48 KB, v>=25 stays zero

  const int ob = blockIdx.x;
  const int n = blockIdx.y;
  const int tid = threadIdx.x;
  const int lane = tid & 31;
  const int wave = tid >> 5;
  const int half = lane >> 4;
  const int l16 = lane & 15;

  for (int i = tid; i < S_ * 16 * TB * 32; i += 256) ((uint16_t*)x3)[i] = 0;

  // Preload all W fragments (A-operand, M=o within this 16-o block), K=64 -> 2 k-steps
  Frag16 wf[4][2];
  {
    int o = ob * 16 + l16;
    for (int s4 = 0; s4 < 4; ++s4) {
      const uint16_t* wp = wbf + ((size_t)s4 * O_ + o) * C_;
#pragma unroll
      for (int kk = 0; kk < 2; ++kk)
#pragma unroll
        for (int j = 0; j < 8; ++j)
          wf[s4][kk].u[j] = *(const uint32_t*)(wp + kk * 32 + kbase(j, half));
    }
  }

  // Hoist loop-invariant scalars: down_b for this lane's 8 output rows,
  // and the ybias sums for this wave's 4 stage-B tasks.
  float dbv[8];
#pragma unroll
  for (int r = 0; r < 8; ++r) dbv[r] = down_b[ob * 16 + r + 8 * half];
  float ybh[4];
#pragma unroll
  for (int q = 0; q < 4; ++q) {
    int task = wave + 8 * q;
    int ol = task >> 1, ut = task & 1;
    int o = ob * 16 + ol;
    int u = ut * 16 + l16;
    float yb = 0.f;
    if (u < V_)
      for (int s = 0; s < S_; ++s)
        yb += ybias[(((size_t)n * S_ + s) * O_ + o) * 32 + u];
    ybh[q] = yb;
  }

  for (int tb = 0; tb < T_ / TB; ++tb) {
    __syncthreads();
    // ---- Stage A: per 16-wide tv column tile ----
    for (int nt = wave; nt < NT; nt += 8) {
      int tvc = nt * 16 + l16;                       // column within this t-block chunk
      const uint16_t* px = xT + (((size_t)n * T_ + (size_t)tb * TB) * V_ + tvc) * C_;
      // prefetch next t-block's column (global_prefetch_b8)
      __builtin_prefetch(px + (size_t)TB * V_ * C_, 0, 0);
      Frag16 b0, b1;                                 // B-operand, N=tv, K=c
#pragma unroll
      for (int j = 0; j < 8; ++j) {
        b0.u[j] = *(const uint32_t*)(px + kbase(j, half));
        b1.u[j] = *(const uint32_t*)(px + 32 + kbase(j, half));
      }
      int tl = tvc / V_;
      int vv = tvc % V_;
      for (int s4 = 0; s4 < 4; ++s4) {
        Acc8 acc;
#pragma unroll
        for (int r = 0; r < 8; ++r) acc.f[r] = 0.f;
        acc.v = __builtin_amdgcn_wmma_f32_16x16x32_bf16(
            false, wf[s4][0].v, false, b0.v, (short)0, acc.v, false, false);
        acc.v = __builtin_amdgcn_wmma_f32_16x16x32_bf16(
            false, wf[s4][1].v, false, b1.v, (short)0, acc.v, false, false);
        if (s4 < S_) {
#pragma unroll
          for (int r = 0; r < 8; ++r)                 // D: M = r + 8*half (= o_local)
            x3[s4][r + 8 * half][tl][vv] = f2bf(acc.f[r]);
        } else {                                      // down path -> xd (bf16) to HBM
#pragma unroll
          for (int r = 0; r < 8; ++r) {
            int o = ob * 16 + r + 8 * half;
            float val = acc.f[r] + dbv[r];
            xdbf[(((size_t)n * O_ + o) * T_ + (tb * TB + tl)) * V_ + vv] = f2bf(val);
          }
        }
      }
    }
    __syncthreads();
    // ---- Stage B: y tile per (o_local, u-tile) ----
#pragma unroll
    for (int q = 0; q < 4; ++q) {
      int task = wave + 8 * q;
      int ol = task >> 1;
      int ut = task & 1;
      int o = ob * 16 + ol;
      Acc8 acc;
#pragma unroll
      for (int r = 0; r < 8; ++r) acc.f[r] = 0.f;
      for (int s = 0; s < S_; ++s) {
        Frag16 a, b;                                  // A: M=t (=l16), K=v ; B: N=u, K=v
#pragma unroll
        for (int j = 0; j < 8; ++j)
          a.u[j] = *(const uint32_t*)&x3[s][ol][l16][kbase(j, half)];
        const uint16_t* ab =
            abf + ((((size_t)n * S_ + s) * O_ + o) * 32 + (ut * 16 + l16)) * 32;
#pragma unroll
        for (int j = 0; j < 8; ++j)
          b.u[j] = *(const uint32_t*)(ab + kbase(j, half));
        acc.v = __builtin_amdgcn_wmma_f32_16x16x32_bf16(
            false, a.v, false, b.v, (short)0, acc.v, false, false);
      }
      int u = ut * 16 + l16;
      if (u < V_) {
        float yb = ybh[q];
        float* yp = yraw + (((size_t)n * O_ + o) * T_ + (size_t)tb * TB) * V_ + u;
#pragma unroll
        for (int r = 0; r < 8; ++r)                   // D: M = r + 8*half (= t_local)
          yp[(size_t)(r + 8 * half) * V_] = acc.f[r] + yb;
      }
    }
  }
}

// ---------------- BN stats (deterministic tree reduction) ----------------
__global__ void k_stats(const float* __restrict__ yraw, const uint16_t* __restrict__ xdbf,
                        float* __restrict__ part) {   // part[8][4][O]
  int o = blockIdx.x, p = blockIdx.y;
  __shared__ float sh[4][256];
  float s0 = 0, s1 = 0, s2 = 0, s3 = 0;
  for (int n = p * 4; n < p * 4 + 4; ++n) {
    size_t base = ((size_t)n * O_ + o) * T_ * V_;
    for (int i = threadIdx.x; i < T_ * V_; i += 256) {
      float yv = yraw[base + i];
      s0 += yv; s1 += yv * yv;
      float xd = bf2f(xdbf[base + i]);
      s2 += xd; s3 += xd * xd;
    }
  }
  sh[0][threadIdx.x] = s0; sh[1][threadIdx.x] = s1;
  sh[2][threadIdx.x] = s2; sh[3][threadIdx.x] = s3;
  __syncthreads();
  for (int st = 128; st > 0; st >>= 1) {
    if (threadIdx.x < st)
      for (int q = 0; q < 4; ++q) sh[q][threadIdx.x] += sh[q][threadIdx.x + st];
    __syncthreads();
  }
  if (threadIdx.x < 4) part[((size_t)p * 4 + threadIdx.x) * O_ + o] = sh[threadIdx.x][0];
}

__global__ void k_coef(const float* __restrict__ part, const float* __restrict__ bnw,
                       const float* __restrict__ bnb, const float* __restrict__ dbnw,
                       const float* __restrict__ dbnb, float* __restrict__ coef) {
  int o = threadIdx.x;
  if (o >= O_) return;
  float s[4] = {0, 0, 0, 0};
  for (int p = 0; p < 8; ++p)
    for (int q = 0; q < 4; ++q) s[q] += part[((size_t)p * 4 + q) * O_ + o];
  const float M = (float)N_ * T_ * V_;
  float m1 = s[0] / M, v1 = s[1] / M - m1 * m1;
  float sc1 = bnw[o] * rsqrtf(v1 + 1e-5f);
  float m2 = s[2] / M, v2 = s[3] / M - m2 * m2;
  float sc2 = dbnw[o] * rsqrtf(v2 + 1e-5f);
  coef[0 * O_ + o] = sc1;
  coef[1 * O_ + o] = bnb[o] - m1 * sc1;
  coef[2 * O_ + o] = sc2;
  coef[3 * O_ + o] = dbnb[o] - m2 * sc2;
}

// out = relu(bn(y) + bn(xd)), in-place on d_out
__global__ void k_final(float* __restrict__ y, const uint16_t* __restrict__ xdbf,
                        const float* __restrict__ coef) {
  size_t i = (size_t)blockIdx.x * 256 + threadIdx.x;
  if (i >= (size_t)N_ * O_ * T_ * V_) return;
  int o = (int)((i / ((size_t)T_ * V_)) % O_);
  float val = coef[0 * O_ + o] * y[i] + coef[1 * O_ + o]
            + coef[2 * O_ + o] * bf2f(xdbf[i]) + coef[3 * O_ + o];
  y[i] = val > 0.f ? val : 0.f;
}

extern "C" void kernel_launch(void* const* d_in, const int* in_sizes, int n_in,
                              void* d_out, int out_size, void* d_ws, size_t ws_size,
                              hipStream_t stream) {
  (void)in_sizes; (void)n_in; (void)out_size; (void)ws_size;
  const float* x     = (const float*)d_in[0];
  const float* spd   = (const float*)d_in[1];
  const float* A3    = (const float*)d_in[2];
  const float* A6    = (const float*)d_in[3];
  const float* c1w   = (const float*)d_in[4];
  const float* c1b   = (const float*)d_in[5];
  const float* c2w   = (const float*)d_in[6];
  const float* c2b   = (const float*)d_in[7];
  const float* c4w   = (const float*)d_in[8];
  const float* c4b   = (const float*)d_in[9];
  const float* c3w   = (const float*)d_in[10];
  const float* c3b   = (const float*)d_in[11];
  const float* alpha = (const float*)d_in[12];
  const float* beta  = (const float*)d_in[13];
  const float* gamma = (const float*)d_in[14];
  const float* bnw   = (const float*)d_in[15];
  const float* bnb   = (const float*)d_in[16];
  const float* dww   = (const float*)d_in[17];
  const float* dwb   = (const float*)d_in[18];
  const float* dbnw  = (const float*)d_in[19];
  const float* dbnb  = (const float*)d_in[20];
  float* y = (float*)d_out;

  char* base = (char*)d_ws;
  size_t off = 0;
  auto alloc = [&](size_t bytes) -> void* {
    void* p = base + off;
    off += (bytes + 255) & ~(size_t)255;
    return p;
  };
  uint16_t* xT    = (uint16_t*)alloc((size_t)N_ * C_ * T_ * V_ * 2);       // 26.2 MB
  uint16_t* abf   = (uint16_t*)alloc((size_t)N_ * S_ * O_ * 32 * 32 * 2);  // 37.7 MB
  uint16_t* xdbf  = (uint16_t*)alloc((size_t)N_ * O_ * T_ * V_ * 2);       // 78.6 MB
  float*    xm    = (float*)alloc((size_t)N_ * C_ * V_ * 4);
  float*    x1    = (float*)alloc((size_t)N_ * S_ * R_ * V_ * 4);
  float*    x2    = (float*)alloc((size_t)N_ * S_ * R_ * V_ * 4);
  float*    dts   = (float*)alloc((size_t)N_ * S_ * R_ * V_ * V_ * 4);
  uint16_t* wbf   = (uint16_t*)alloc((size_t)4 * O_ * C_ * 2);
  float*    ybias = (float*)alloc((size_t)N_ * S_ * O_ * 32 * 4);
  float*    part  = (float*)alloc((size_t)8 * 4 * O_ * 4);
  float*    coef  = (float*)alloc((size_t)4 * O_ * 4);

  int tot;
  tot = N_ * C_ * T_ * V_;
  k_cvt_xT<<<(tot + 255) / 256, 256, 0, stream>>>(x, xT);
  tot = N_ * C_ * V_;
  k_mean_t<<<(tot + 255) / 256, 256, 0, stream>>>(x, xm);
  tot = 2 * N_ * S_ * R_ * V_;
  k_x12<<<(tot + 255) / 256, 256, 0, stream>>>(xm, c1w, c1b, c2w, c2b, x1, x2);
  tot = N_ * S_ * R_ * V_ * V_;
  k_dtanh<<<(tot + 255) / 256, 256, 0, stream>>>(x1, x2, dts);
  tot = 4 * O_ * C_;
  k_wpack<<<(tot + 255) / 256, 256, 0, stream>>>(c3w, dww, wbf);
  k_abf<<<N_ * S_ * O_, 256, 0, stream>>>(dts, c4w, c4b, A3, A6, spd,
                                          alpha, beta, gamma, c3b, abf, ybias);
  k_main<<<dim3(O_ / 16, N_), 256, 0, stream>>>(xT, wbf, abf, ybias, dwb, xdbf, y);
  k_stats<<<dim3(O_, 8), 256, 0, stream>>>(y, xdbf, part);
  k_coef<<<1, 256, 0, stream>>>(part, bnw, bnb, dbnw, dbnb, coef);
  tot = N_ * O_ * T_ * V_;
  k_final<<<(tot + 255) / 256, 256, 0, stream>>>(y, xdbf, coef);
}
